// MinkUNetBase_1675037246165
// MI455X (gfx1250) — matrix operand order repaired and checked by reading, not compiled, not run on hardware
//
#include <hip/hip_runtime.h>

// ---------------------------------------------------------------------------
// MinkUNet forward for gfx1250 (MI455X): gather-WMMA-scatter sparse convs.
// GEMMs run on v_wmma_f32_16x16x32_f16 (f16 in, f32 accumulate), scatter via
// global f32 atomics, BN via LDS reduction + fused apply.
// sconv: B tile staged in LDS (shared by all 4 waves of the block), A rows
// fetched as float4, 2 M-tiles per wave -> 2 back-to-back WMMAs per K-chunk.
// MODE 0: Cin == 4 (conv0, single guard-free float4 per row, zero-padded).
// MODE 1: Cin % 32 == 0 (all other convs).
// ---------------------------------------------------------------------------

typedef __attribute__((ext_vector_type(16))) _Float16 v16h;
typedef __attribute__((ext_vector_type(8)))  float    v8f;

#define WPB 4   // waves per block (wave32)

// ------------------------------ fill ---------------------------------------
__global__ void fill0_k(float* __restrict__ p, long n) {
  long i = (long)blockIdx.x * blockDim.x + threadIdx.x;
  long stride = (long)gridDim.x * blockDim.x;
  for (; i < n; i += stride) p[i] = 0.0f;
}

// ------------------------- sparse conv (gather-WMMA-scatter) ---------------
// x [Nin,Cin] f32, W [K,Cin,Cout] f32, im/om [K,Mpad] int32, acc [nOut,Cout].
// One wave computes two 16(M) x 16(Cout) tiles (32 rows) for kernel-offset k.
template <int MODE>   // 0: Cin==4, 1: Cin%32==0
__global__ void __launch_bounds__(32 * WPB)
sconv_wmma(const float* __restrict__ x, const float* __restrict__ W,
           const int* __restrict__ im, const int* __restrict__ om,
           const int* __restrict__ nOutPtr, float* __restrict__ acc,
           int Mpad, int Cin, int Cout)
{
  // B tile staged transposed in LDS: Bt[col 0..15][k-in-chunk 0..31] (+pad)
  __shared__ float Bt[16][36];

  const int k    = blockIdx.x;
  const int tid  = threadIdx.x;
  const int lane = tid & 31;
  const int wave = tid >> 5;
  const int mbase  = (blockIdx.y * WPB + wave) * 32;
  const bool active = mbase < Mpad;      // inactive waves still run barriers+WMMA
  const int  nOut = nOutPtr[0];
  const int  hl   = lane & 15;
  const bool hi   = lane >= 16;
  const int  colBase = blockIdx.z * 16;
  const int  col     = colBase + hl;

  // A gather: lane holds row M = hl of tile0 and tile1 (tile1 = +16 rows)
  const int r0 = mbase + hl;
  const int r1 = mbase + 16 + hl;
  const int g0 = (active && r0 < Mpad) ? im[k * Mpad + r0] : 0;
  const int g1 = (active && r1 < Mpad) ? im[k * Mpad + r1] : 0;
  const float* __restrict__ xr0 = x + (long)g0 * Cin;
  const float* __restrict__ xr1 = x + (long)g1 * Cin;
  const float* __restrict__ Wk  = W + (long)k * Cin * Cout;

  // ISA 16-bit layouts (wave32): A lanes0-15 K=0..7/16..23, lanes16-31 K=8..15/24..31
  //                              B lanes0-15 K=0..7/16..23, lanes16-31 K=8..15/24..31 (same as A)
  const int kbA = hi ? 8 : 0;
  const int kbB = hi ? 16 : 0;

  // cooperative-load coordinates: 128 threads cover 32 rows x 16 cols (float4)
  const int rr = tid >> 2;        // 0..31 : K within chunk
  const int qq = (tid & 3) * 4;   // 0,4,8,12 : col quad

  const int nChunks = (MODE == 0) ? 1 : (Cin >> 5);

  v8f c0 = {}, c1 = {};
  for (int chk = 0; chk < nChunks; ++chk) {
    const int cc = chk << 5;

    // ---- stage B chunk (32x16) into LDS, transposed ----
    float4 w = make_float4(0.f, 0.f, 0.f, 0.f);
    if (MODE == 1 || rr < Cin)
      w = *(const float4*)(Wk + (long)(cc + rr) * Cout + colBase + qq);
    Bt[qq + 0][rr] = w.x;
    Bt[qq + 1][rr] = w.y;
    Bt[qq + 2][rr] = w.z;
    Bt[qq + 3][rr] = w.w;
    if (MODE == 1 && chk + 1 < nChunks)  // hide next weight fetch latency
      __builtin_prefetch(Wk + (long)(cc + 32 + rr) * Cout + colBase + qq, 0, 0);
    __syncthreads();

    // ---- read per-lane B column (contiguous in LDS) ----
    v16h b;
    {
      const float* brow = &Bt[hl][kbA];
#pragma unroll
      for (int j = 0; j < 16; ++j) b[j] = (_Float16)brow[j + (j >= 8 ? 8 : 0)];
    }

    // ---- fetch A rows ----
    v16h a0, a1;
    if (MODE == 0) {
      // Cin == 4: one float4 per row; K>=4 are zero; hi lanes (K>=8) all zero
      const _Float16 z = (_Float16)0.0f;
      float4 f0 = *(const float4*)xr0;
      float4 f1 = *(const float4*)xr1;
      a0[0] = hi ? z : (_Float16)f0.x;  a0[1] = hi ? z : (_Float16)f0.y;
      a0[2] = hi ? z : (_Float16)f0.z;  a0[3] = hi ? z : (_Float16)f0.w;
      a1[0] = hi ? z : (_Float16)f1.x;  a1[1] = hi ? z : (_Float16)f1.y;
      a1[2] = hi ? z : (_Float16)f1.z;  a1[3] = hi ? z : (_Float16)f1.w;
#pragma unroll
      for (int j = 4; j < 16; ++j) { a0[j] = z; a1[j] = z; }
    } else {
      float4 p0 = *(const float4*)(xr0 + cc + kbA);
      float4 p1 = *(const float4*)(xr0 + cc + kbA + 4);
      float4 p2 = *(const float4*)(xr0 + cc + 16 + kbA);
      float4 p3 = *(const float4*)(xr0 + cc + 16 + kbA + 4);
      float4 q0 = *(const float4*)(xr1 + cc + kbA);
      float4 q1 = *(const float4*)(xr1 + cc + kbA + 4);
      float4 q2 = *(const float4*)(xr1 + cc + 16 + kbA);
      float4 q3 = *(const float4*)(xr1 + cc + 16 + kbA + 4);
      a0[0]  = (_Float16)p0.x; a0[1]  = (_Float16)p0.y;
      a0[2]  = (_Float16)p0.z; a0[3]  = (_Float16)p0.w;
      a0[4]  = (_Float16)p1.x; a0[5]  = (_Float16)p1.y;
      a0[6]  = (_Float16)p1.z; a0[7]  = (_Float16)p1.w;
      a0[8]  = (_Float16)p2.x; a0[9]  = (_Float16)p2.y;
      a0[10] = (_Float16)p2.z; a0[11] = (_Float16)p2.w;
      a0[12] = (_Float16)p3.x; a0[13] = (_Float16)p3.y;
      a0[14] = (_Float16)p3.z; a0[15] = (_Float16)p3.w;
      a1[0]  = (_Float16)q0.x; a1[1]  = (_Float16)q0.y;
      a1[2]  = (_Float16)q0.z; a1[3]  = (_Float16)q0.w;
      a1[4]  = (_Float16)q1.x; a1[5]  = (_Float16)q1.y;
      a1[6]  = (_Float16)q1.z; a1[7]  = (_Float16)q1.w;
      a1[8]  = (_Float16)q2.x; a1[9]  = (_Float16)q2.y;
      a1[10] = (_Float16)q2.z; a1[11] = (_Float16)q2.w;
      a1[12] = (_Float16)q3.x; a1[13] = (_Float16)q3.y;
      a1[14] = (_Float16)q3.z; a1[15] = (_Float16)q3.w;
    }

    // EXEC is all-ones here (required by WMMA); inactive waves compute garbage
    // they never scatter.
    c0 = __builtin_amdgcn_wmma_f32_16x16x32_f16(false, a0, false, b,
                                                (short)0, c0, false, false);
    c1 = __builtin_amdgcn_wmma_f32_16x16x32_f16(false, a1, false, b,
                                                (short)0, c1, false, false);
    __syncthreads();   // protect Bt before next chunk's store
  }

  // Scatter-add: C VGPR r -> M = r (+8 for hi lanes), N = hl
  if (active) {
#pragma unroll
    for (int r = 0; r < 8; ++r) {
      int m = mbase + r + (hi ? 8 : 0);
      if (m < Mpad) {
        int oi = om[k * Mpad + m];
        if (oi < nOut)                     // oi == nOut marks padded slot
          atomicAdd(&acc[(long)oi * Cout + col], c0[r]);
      }
      int m2 = m + 16;
      if (m2 < Mpad) {
        int oi = om[k * Mpad + m2];
        if (oi < nOut)
          atomicAdd(&acc[(long)oi * Cout + col], c1[r]);
      }
    }
  }
}

// ------------------------- dense GEMM (1x1 downsample) ---------------------
// All downsample Cin are multiples of 32.
__global__ void __launch_bounds__(32 * WPB)
gemm_wmma(const float* __restrict__ x, const float* __restrict__ W,
          const int* __restrict__ nPtr, float* __restrict__ out,
          int Cin, int Cout)
{
  const int lane  = threadIdx.x & 31;
  const int wave  = threadIdx.x >> 5;
  const int mbase = (blockIdx.x * WPB + wave) * 16;
  const int n     = nPtr[0];
  if (mbase >= n) return;                 // wave-uniform
  const int  hl  = lane & 15;
  const bool hi  = lane >= 16;
  const int  col = blockIdx.y * 16 + hl;
  const int  arow = mbase + hl;
  const bool av   = arow < n;
  const float* __restrict__ xrow = x + (long)(av ? arow : 0) * Cin;
  const int kbA = hi ? 8 : 0;
  const int kbB = hi ? 16 : 0;

  v8f c = {};
  for (int c0 = 0; c0 < Cin; c0 += 32) {
    v16h a, b;
    float4 p0 = *(const float4*)(xrow + c0 + kbA);
    float4 p1 = *(const float4*)(xrow + c0 + kbA + 4);
    float4 p2 = *(const float4*)(xrow + c0 + 16 + kbA);
    float4 p3 = *(const float4*)(xrow + c0 + 16 + kbA + 4);
    a[0]  = (_Float16)p0.x; a[1]  = (_Float16)p0.y;
    a[2]  = (_Float16)p0.z; a[3]  = (_Float16)p0.w;
    a[4]  = (_Float16)p1.x; a[5]  = (_Float16)p1.y;
    a[6]  = (_Float16)p1.z; a[7]  = (_Float16)p1.w;
    a[8]  = (_Float16)p2.x; a[9]  = (_Float16)p2.y;
    a[10] = (_Float16)p2.z; a[11] = (_Float16)p2.w;
    a[12] = (_Float16)p3.x; a[13] = (_Float16)p3.y;
    a[14] = (_Float16)p3.z; a[15] = (_Float16)p3.w;
#pragma unroll
    for (int j = 0; j < 16; ++j)
      b[j] = (_Float16)W[(long)(c0 + kbA + j + (j >= 8 ? 8 : 0)) * Cout + col];
    c = __builtin_amdgcn_wmma_f32_16x16x32_f16(false, a, false, b,
                                               (short)0, c, false, false);
  }
#pragma unroll
  for (int r = 0; r < 8; ++r) {
    int m = mbase + r + (hi ? 8 : 0);
    if (m < n) out[(long)m * Cout + col] = c[r];
  }
}

// ------------------------------ BatchNorm ----------------------------------
__global__ void bn_stats_k(const float* __restrict__ x, const int* __restrict__ nPtr,
                           int C, float* __restrict__ stats /*[2C] sum,sumsq*/)
{
  __shared__ float ls[256], lq[256];
  const int ch = blockIdx.y;
  const int n  = nPtr[0];
  float s = 0.0f, q = 0.0f;
  for (long r = (long)blockIdx.x * blockDim.x + threadIdx.x; r < n;
       r += (long)gridDim.x * blockDim.x) {
    float v = x[r * C + ch];
    s += v; q += v * v;
  }
  ls[threadIdx.x] = s; lq[threadIdx.x] = q;
  __syncthreads();
  for (int off = 128; off > 0; off >>= 1) {
    if ((int)threadIdx.x < off) {
      ls[threadIdx.x] += ls[threadIdx.x + off];
      lq[threadIdx.x] += lq[threadIdx.x + off];
    }
    __syncthreads();
  }
  if (threadIdx.x == 0) {
    atomicAdd(&stats[ch],     ls[0]);
    atomicAdd(&stats[C + ch], lq[0]);
  }
}

__global__ void bn_apply_k(const float* __restrict__ x, const float* __restrict__ stats,
                           const float* __restrict__ g, const float* __restrict__ b,
                           const int* __restrict__ nPtr, int C,
                           float* __restrict__ y, const float* __restrict__ resid,
                           int doRelu)
{
  const int  n     = nPtr[0];
  const long total = (long)n * C;
  const float fn   = (float)n;
  long i = (long)blockIdx.x * blockDim.x + threadIdx.x;
  long stride = (long)gridDim.x * blockDim.x;
  for (; i < total; i += stride) {
    int ch = (int)(i % C);
    float m = stats[ch] / fn;
    float v = stats[C + ch] / fn - m * m;           // biased var, as reference
    float val = (x[i] - m) * rsqrtf(v + 1e-5f) * g[ch] + b[ch];
    if (resid)  val += resid[i];
    if (doRelu) val = fmaxf(val, 0.0f);
    y[i] = val;
  }
}

// ------------------------------ concat -------------------------------------
__global__ void concat2_k(const float* __restrict__ a, int Ca,
                          const float* __restrict__ b, int Cb,
                          const int* __restrict__ nPtr, float* __restrict__ out)
{
  const int n = nPtr[0];
  const int C = Ca + Cb;
  const long total = (long)n * C;
  long i = (long)blockIdx.x * blockDim.x + threadIdx.x;
  long stride = (long)gridDim.x * blockDim.x;
  for (; i < total; i += stride) {
    int row = (int)(i / C);
    int ch  = (int)(i % C);
    out[i] = (ch < Ca) ? a[(long)row * Ca + ch] : b[(long)row * Cb + (ch - Ca)];
  }
}

// ---------------------------------------------------------------------------
// Host orchestration. Input flattening assumption (dict creation order):
//   0: x [n0,4]
//   1..96: params in creation order:
//     1 conv0, 2 bn0g, 3 bn0b, 4 conv1, 5 bn1g, 6 bn1b, 7 conv2, 8 bn2g, 9 bn2b,
//     10 conv3, 11 bn3g, 12 bn3b, 13 conv4, 14 bn4g, 15 bn4b,
//     16-21 b1{c1,c2,bn1g,bn1b,bn2g,bn2b},
//     22-30 b2{c1,c2,bn1g,bn1b,bn2g,bn2b,ds,dsbng,dsbnb},
//     31-39 b3{...}, 40-48 b4{...},
//     49-51 tr4,btr4g,btr4b, 52-60 b5{...}, 61-63 tr5..., 64-72 b6{...},
//     73-75 tr6..., 76-84 b7{...}, 85-87 tr7..., 88-96 b8{...}
//   97..101: n0..n4 (scalars), 102-103 km5_0(im,om), 104-113 km3_0..km3_4,
//   114-129: kd0,kt0,kd1,kt1,kd2,kt2,kd3,kt3 (im,om each).
// ---------------------------------------------------------------------------
extern "C" void kernel_launch(void* const* d_in, const int* in_sizes, int n_in,
                              void* d_out, int out_size, void* d_ws, size_t ws_size,
                              hipStream_t stream)
{
  (void)n_in; (void)out_size; (void)ws_size;
  auto F = [&](int i) { return (const float*)d_in[i]; };
  auto I = [&](int i) { return (const int*)d_in[i]; };

  const int N0  = in_sizes[0] / 4;        // actual n0
  const int N1B = 32 * 32 * 32;           // hard coord-range bounds for n1..n4
  const int N2B = 16 * 16 * 16;
  const int N3B = 8 * 8 * 8;
  const int N4B = 4 * 4 * 4;

  const int Mp_km5_0 = in_sizes[102] / 125;
  const int Mp_km3_0 = in_sizes[104] / 27;
  const int Mp_km3_1 = in_sizes[106] / 27;
  const int Mp_km3_2 = in_sizes[108] / 27;
  const int Mp_km3_3 = in_sizes[110] / 27;
  const int Mp_km3_4 = in_sizes[112] / 27;
  const int Mp_kd0 = in_sizes[114] / 8, Mp_kt0 = in_sizes[116] / 8;
  const int Mp_kd1 = in_sizes[118] / 8, Mp_kt1 = in_sizes[120] / 8;
  const int Mp_kd2 = in_sizes[122] / 8, Mp_kt2 = in_sizes[124] / 8;
  const int Mp_kd3 = in_sizes[126] / 8, Mp_kt3 = in_sizes[128] / 8;

  const int* n0p = I(97);
  const int* n1p = I(98);
  const int* n2p = I(99);
  const int* n3p = I(100);
  const int* n4p = I(101);

  // ---- bump allocator over workspace ----
  char* wp = (char*)d_ws;
  auto alloc = [&](size_t elems) -> float* {
    float* p = (float*)wp;
    wp += ((elems * sizeof(float)) + 255) & ~(size_t)255;
    return p;
  };
  float* scrA  = alloc((size_t)N0 * 96);   // conv accumulator
  float* scrB  = alloc((size_t)N0 * 96);   // block h1
  float* scrC  = alloc((size_t)N0 * 96);   // block identity (downsample)
  float* catB  = alloc((size_t)N0 * 128);  // concat buffer
  float* stg1  = alloc((size_t)N0 * 96);   // stage ping
  float* stg2  = alloc((size_t)N0 * 96);   // stage pong
  float* op1   = alloc((size_t)N0 * 32);   // skip L0
  float* ob1   = alloc((size_t)N1B * 32);  // skip L1
  float* ob2   = alloc((size_t)N2B * 64);  // skip L2
  float* ob3   = alloc((size_t)N3B * 128); // skip L3
  float* stats = alloc(1024);              // BN sum/sumsq (max C=384)

  // ---- launch helpers (all on `stream`, capture-safe) ----
  auto fill0 = [&](float* p, long n) {
    int blocks = (int)((n + 255) / 256);
    fill0_k<<<blocks, 256, 0, stream>>>(p, n);
  };
  auto sconv = [&](const float* xin, const float* W, int K,
                   const int* im, const int* om, int Mpad,
                   const int* nPtr, int nBound, int Cin, int Cout, float* acc) {
    fill0(acc, (long)nBound * Cout);
    dim3 grid(K, (Mpad + 32 * WPB - 1) / (32 * WPB), Cout / 16);
    if (Cin == 4)
      sconv_wmma<0><<<grid, 32 * WPB, 0, stream>>>(xin, W, im, om, nPtr, acc,
                                                   Mpad, Cin, Cout);
    else   // all remaining Cin in this net are multiples of 32
      sconv_wmma<1><<<grid, 32 * WPB, 0, stream>>>(xin, W, im, om, nPtr, acc,
                                                   Mpad, Cin, Cout);
  };
  auto bn = [&](const float* xin, const float* g, const float* bb,
                const int* nPtr, int nBound, int C, float* y,
                const float* resid, int relu) {
    fill0(stats, 2 * C);
    bn_stats_k<<<dim3(64, C), 256, 0, stream>>>(xin, nPtr, C, stats);
    long total = (long)nBound * C;
    int blocks = (int)((total + 255) / 256);
    bn_apply_k<<<blocks, 256, 0, stream>>>(xin, stats, g, bb, nPtr, C, y, resid, relu);
  };
  auto gemm = [&](const float* xin, const float* W, const int* nPtr, int nBound,
                  int Cin, int Cout, float* out) {
    dim3 grid((nBound + 16 * WPB - 1) / (16 * WPB), Cout / 16);
    gemm_wmma<<<grid, 32 * WPB, 0, stream>>>(xin, W, nPtr, out, Cin, Cout);
  };
  auto concat = [&](const float* a, int Ca, const float* b, int Cb,
                    const int* nPtr, int nBound, float* out) {
    long total = (long)nBound * (Ca + Cb);
    int blocks = (int)((total + 255) / 256);
    concat2_k<<<blocks, 256, 0, stream>>>(a, Ca, b, Cb, nPtr, out);
  };
  // ResNet BasicBlock: relu(bn(sconv(x,c1)))->h1; bn(sconv(h1,c2)) + idn -> relu
  auto blockf = [&](const float* xin, int Cin, int Cpl,
                    const float* c1, const float* c2,
                    const float* g1, const float* b1,
                    const float* g2, const float* b2,
                    const float* ds, const float* dsg, const float* dsb,
                    const int* im, const int* om, int Mpad,
                    const int* nPtr, int nBound, float* out) {
    sconv(xin, c1, 27, im, om, Mpad, nPtr, nBound, Cin, Cpl, scrA);
    bn(scrA, g1, b1, nPtr, nBound, Cpl, scrB, nullptr, 1);
    sconv(scrB, c2, 27, im, om, Mpad, nPtr, nBound, Cpl, Cpl, scrA);
    const float* resid = xin;
    if (ds) {
      gemm(xin, ds, nPtr, nBound, Cin, Cpl, scrC);
      bn(scrC, dsg, dsb, nPtr, nBound, Cpl, scrC, nullptr, 0);  // in-place ok
      resid = scrC;
    }
    bn(scrA, g2, b2, nPtr, nBound, Cpl, out, resid, 1);
  };

  const float* X = F(0);

  // op1 = relu(bn(sconv(x, conv0, km5_0)))           [N0,32]
  sconv(X, F(1), 125, I(102), I(103), Mp_km5_0, n0p, N0, 4, 32, scrA);
  bn(scrA, F(2), F(3), n0p, N0, 32, op1, nullptr, 1);

  // level1 = relu(bn(sconv(op1, conv1, kd0)))        [N1,32]
  sconv(op1, F(4), 8, I(114), I(115), Mp_kd0, n1p, N1B, 32, 32, scrA);
  bn(scrA, F(5), F(6), n1p, N1B, 32, stg1, nullptr, 1);
  // ob1 = block b1 (32->32, km3_1)
  blockf(stg1, 32, 32, F(16), F(17), F(18), F(19), F(20), F(21),
         nullptr, nullptr, nullptr, I(106), I(107), Mp_km3_1, n1p, N1B, ob1);

  // level2 = relu(bn(sconv(ob1, conv2, kd1)))        [N2,32]
  sconv(ob1, F(7), 8, I(118), I(119), Mp_kd1, n2p, N2B, 32, 32, scrA);
  bn(scrA, F(8), F(9), n2p, N2B, 32, stg1, nullptr, 1);
  // ob2 = block b2 (32->64, ds, km3_2)
  blockf(stg1, 32, 64, F(22), F(23), F(24), F(25), F(26), F(27),
         F(28), F(29), F(30), I(108), I(109), Mp_km3_2, n2p, N2B, ob2);

  // level3 = relu(bn(sconv(ob2, conv3, kd2)))        [N3,64]
  sconv(ob2, F(10), 8, I(122), I(123), Mp_kd2, n3p, N3B, 64, 64, scrA);
  bn(scrA, F(11), F(12), n3p, N3B, 64, stg1, nullptr, 1);
  // ob3 = block b3 (64->128, ds, km3_3)
  blockf(stg1, 64, 128, F(31), F(32), F(33), F(34), F(35), F(36),
         F(37), F(38), F(39), I(110), I(111), Mp_km3_3, n3p, N3B, ob3);

  // level4 = relu(bn(sconv(ob3, conv4, kd3)))        [N4,128]
  sconv(ob3, F(13), 8, I(126), I(127), Mp_kd3, n4p, N4B, 128, 128, scrA);
  bn(scrA, F(14), F(15), n4p, N4B, 128, stg1, nullptr, 1);
  // ob4 = block b4 (128->256, ds, km3_4) -> stg2
  blockf(stg1, 128, 256, F(40), F(41), F(42), F(43), F(44), F(45),
         F(46), F(47), F(48), I(112), I(113), Mp_km3_4, n4p, N4B, stg2);

  // tr4 (kt3): L4 -> L3                              [N3,256]
  sconv(stg2, F(49), 8, I(128), I(129), Mp_kt3, n3p, N3B, 256, 256, scrA);
  bn(scrA, F(50), F(51), n3p, N3B, 256, stg1, nullptr, 1);
  // b5 on concat([N3,256],[N3,128]) -> 256
  concat(stg1, 256, ob3, 128, n3p, N3B, catB);
  blockf(catB, 384, 256, F(52), F(53), F(54), F(55), F(56), F(57),
         F(58), F(59), F(60), I(110), I(111), Mp_km3_3, n3p, N3B, stg2);

  // tr5 (kt2): L3 -> L2                              [N2,128]
  sconv(stg2, F(61), 8, I(124), I(125), Mp_kt2, n2p, N2B, 256, 128, scrA);
  bn(scrA, F(62), F(63), n2p, N2B, 128, stg1, nullptr, 1);
  concat(stg1, 128, ob2, 64, n2p, N2B, catB);
  blockf(catB, 192, 128, F(64), F(65), F(66), F(67), F(68), F(69),
         F(70), F(71), F(72), I(108), I(109), Mp_km3_2, n2p, N2B, stg2);

  // tr6 (kt1): L2 -> L1                              [N1,96]
  sconv(stg2, F(73), 8, I(120), I(121), Mp_kt1, n1p, N1B, 128, 96, scrA);
  bn(scrA, F(74), F(75), n1p, N1B, 96, stg1, nullptr, 1);
  concat(stg1, 96, ob1, 32, n1p, N1B, catB);
  blockf(catB, 128, 96, F(76), F(77), F(78), F(79), F(80), F(81),
         F(82), F(83), F(84), I(106), I(107), Mp_km3_1, n1p, N1B, stg2);

  // tr7 (kt0): L1 -> L0                              [N0,96]
  sconv(stg2, F(85), 8, I(116), I(117), Mp_kt0, n0p, N0, 96, 96, scrA);
  bn(scrA, F(86), F(87), n0p, N0, 96, stg1, nullptr, 1);
  concat(stg1, 96, op1, 32, n0p, N0, catB);
  // b8 writes final [n0,96] straight into d_out
  blockf(catB, 128, 96, F(88), F(89), F(90), F(91), F(92), F(93),
         F(94), F(95), F(96), I(104), I(105), Mp_km3_0, n0p, N0, (float*)d_out);
}
